// GEARS_Model_17678085390437
// MI455X (gfx1250) — compile-verified, matrix operand order checked by reading, not run on hardware
//
#include <hip/hip_runtime.h>
#include <hip/hip_bf16.h>

typedef __attribute__((ext_vector_type(16))) _Float16 v16h;
typedef __attribute__((ext_vector_type(8)))  _Float16 v8h;
typedef __attribute__((ext_vector_type(8)))  float    v8f;

#define DHID 64
#define CDIM 128
#define BN_EPS 1e-5f

// ---------------------------------------------------------------------------
// 1. Prep: Wc = W_sg @ W1 (folded weight), store transposed as f16 [128][64];
//    bc = b_sg @ W1 ; zero the BN-stat accumulators. One block, 128 threads.
// ---------------------------------------------------------------------------
__global__ void prep_kernel(const float* __restrict__ Wsg,   // [64,64]
                            const float* __restrict__ W1,    // [64,128]
                            const float* __restrict__ bsg,   // [64]
                            _Float16* __restrict__ wt,       // [128,64] = Wc^T f16
                            float* __restrict__ bc,          // [128]
                            float* __restrict__ S1, float* __restrict__ S2) {
  int c = threadIdx.x;               // 0..127  (output column)
  float b = 0.f;
  for (int j = 0; j < DHID; ++j) b += bsg[j] * W1[j * CDIM + c];
  bc[c] = b;
  for (int i = 0; i < DHID; ++i) {
    float s = 0.f;
    for (int j = 0; j < DHID; ++j) s += Wsg[i * DHID + j] * W1[j * CDIM + c];
    wt[c * DHID + i] = (_Float16)s;
  }
  S1[c] = 0.f;
  S2[c] = 0.f;
}

// ---------------------------------------------------------------------------
// 2. Degree: deg[n] = 1 (self loop) + #incoming edges, then dinv = rsqrt(deg)
// ---------------------------------------------------------------------------
__global__ void deg_init_kernel(float* __restrict__ deg, int N) {
  int n = blockIdx.x * blockDim.x + threadIdx.x;
  if (n < N) deg[n] = 1.0f;
}

__global__ void deg_count_kernel(const int* __restrict__ dst, float* __restrict__ deg, int E) {
  int e = blockIdx.x * blockDim.x + threadIdx.x;
  if (e < E) atomicAdd(&deg[dst[e]], 1.0f);
}

__global__ void dinv_kernel(float* __restrict__ deg_dinv, float* __restrict__ selfw, int N) {
  int n = blockIdx.x * blockDim.x + threadIdx.x;
  if (n < N) {
    float dv = rsqrtf(deg_dinv[n]);
    deg_dinv[n] = dv;          // in-place: becomes dinv
    selfw[n] = dv * dv;
  }
}

// ---------------------------------------------------------------------------
// 3. SGConv hop: hout = hin * selfw  (init), then scatter-add over edges.
//    One wave per edge in the scatter: lanes cover the 64 channels (2 each),
//    so reads of hin[src,*] are fully coalesced 128B lines.
// ---------------------------------------------------------------------------
__global__ void selfmul_kernel(const float* __restrict__ hin,
                               const float* __restrict__ selfw,
                               float* __restrict__ hout, int total) {
  int i = blockIdx.x * blockDim.x + threadIdx.x;
  if (i < total) hout[i] = hin[i] * selfw[i >> 6];
}

__global__ void scatter_kernel(const int* __restrict__ src, const int* __restrict__ dst,
                               const float* __restrict__ dinv,
                               const float* __restrict__ hin, float* __restrict__ hout, int E) {
  int gid = blockIdx.x * blockDim.x + threadIdx.x;
  int e = gid >> 5;
  int lane = gid & 31;
  if (e >= E) return;
  int s = src[e], d = dst[e];
  float ew = dinv[s] * dinv[d];
  float v0 = hin[s * DHID + lane] * ew;
  float v1 = hin[s * DHID + 32 + lane] * ew;
  atomicAdd(&hout[d * DHID + lane], v0);
  atomicAdd(&hout[d * DHID + 32 + lane], v1);
}

// ---------------------------------------------------------------------------
// 4. f32 -> f16 for the WMMA A-matrix
// ---------------------------------------------------------------------------
__global__ void tof16_kernel(const float* __restrict__ in, _Float16* __restrict__ out, int total) {
  int i = blockIdx.x * blockDim.x + threadIdx.x;
  if (i < total) out[i] = (_Float16)in[i];
}

// ---------------------------------------------------------------------------
// 5. WMMA GEMM: G[N,128] = h16[N,64] @ Wc[64,128] + bc.
//    One wave per 16x16 output tile; K=64 -> two v_wmma_f32_16x16x32_f16.
//    A fragment (ISA 16-bit A layout): lane<16 holds K k0+{0..7,16..23} of row
//    M=lane; lane>=16 holds K k0+{8..15,24..31}. B from WcT mirrors A.
//    C/D layout: VGPR r -> row 16*tM + r + 8*(lane>=16), col = lane&15.
// ---------------------------------------------------------------------------
__global__ void gemm_wmma_kernel(const _Float16* __restrict__ h16,  // [N,64]
                                 const _Float16* __restrict__ wt,   // [128,64]
                                 const float* __restrict__ bc,      // [128]
                                 float* __restrict__ G, int N) {
  int wave = (blockIdx.x * blockDim.x + threadIdx.x) >> 5;
  int lane = threadIdx.x & 31;
  int tM = wave >> 3;                 // CDIM/16 = 8 tiles along columns
  int tN = wave & 7;
  if (tM * 16 >= N) return;
  int half = lane >> 4;
  int l15 = lane & 15;
  int m = tM * 16 + l15;
  int n = tN * 16 + l15;

  v8f c = {};
#pragma unroll
  for (int k0 = 0; k0 < DHID; k0 += 32) {
    v8h alo = *(const v8h*)(h16 + (size_t)m * DHID + k0 + 8 * half);
    v8h ahi = *(const v8h*)(h16 + (size_t)m * DHID + k0 + 16 + 8 * half);
    v8h blo = *(const v8h*)(wt + n * DHID + k0 + 8 * half);
    v8h bhi = *(const v8h*)(wt + n * DHID + k0 + 16 + 8 * half);
    v16h a, b;
#pragma unroll
    for (int i = 0; i < 8; ++i) {
      a[i] = alo[i]; a[i + 8] = ahi[i];
      b[i] = blo[i]; b[i + 8] = bhi[i];
    }
    c = __builtin_amdgcn_wmma_f32_16x16x32_f16(false, a, false, b, (short)0, c, false, false);
  }

  int row0 = tM * 16 + 8 * half;
  int col = tN * 16 + l15;
  float bcv = bc[col];
#pragma unroll
  for (int r = 0; r < 8; ++r)
    G[(size_t)(row0 + r) * CDIM + col] = c[r] + bcv;
}

// ---------------------------------------------------------------------------
// 6. Column sums of x over the batch: Sx[n], Sx2[n]
// ---------------------------------------------------------------------------
__global__ void sx_kernel(const float* __restrict__ x, float* __restrict__ Sx,
                          float* __restrict__ Sx2, int N, int B) {
  int n = blockIdx.x * blockDim.x + threadIdx.x;
  if (n >= N) return;
  float s = 0.f, s2 = 0.f;
  for (int b = 0; b < B; ++b) {
    float v = x[(size_t)b * N + n];
    s += v; s2 += v * v;
  }
  Sx[n] = s;
  Sx2[n] = s2;
}

// ---------------------------------------------------------------------------
// 7. BN-stat reduction: S1[c] = sum_n G[n,c]*Sx[n] ; S2[c] = sum_n G^2*Sx2[n]
//    128 threads/block (thread = column c, coalesced across G rows).
// ---------------------------------------------------------------------------
__global__ void reduce_kernel(const float* __restrict__ G, const float* __restrict__ Sx,
                              const float* __restrict__ Sx2,
                              float* __restrict__ S1, float* __restrict__ S2, int N) {
  int c = threadIdx.x;  // 0..127
  int chunk = (N + gridDim.x - 1) / gridDim.x;
  int n0 = blockIdx.x * chunk;
  int n1 = n0 + chunk; if (n1 > N) n1 = N;
  float s1 = 0.f, s2 = 0.f;
  for (int n = n0; n < n1; ++n) {
    float g = G[(size_t)n * CDIM + c];
    s1 += g * Sx[n];
    s2 += g * g * Sx2[n];
  }
  atomicAdd(&S1[c], s1);
  atomicAdd(&S2[c], s2);
}

// ---------------------------------------------------------------------------
// 8. BN coefficients: a = gamma*rsqrt(var+eps), q = beta - a*mean(x*G part)
// ---------------------------------------------------------------------------
__global__ void coef_kernel(const float* __restrict__ S1, const float* __restrict__ S2,
                            const float* __restrict__ gamma, const float* __restrict__ beta,
                            float* __restrict__ A, float* __restrict__ Q, float invBN) {
  int c = threadIdx.x;
  float m = S1[c] * invBN;                 // mean of the x*G term (b1 shift cancels in var)
  float var = S2[c] * invBN - m * m;
  float a = gamma[c] * rsqrtf(var + BN_EPS);
  A[c] = a;
  Q[c] = beta[c] - a * m;                  // a*(b1 - (m+b1)) + beta
}

__global__ void scaleP_kernel(const float* __restrict__ G, const float* __restrict__ A,
                              float* __restrict__ P, int total) {
  int i = blockIdx.x * blockDim.x + threadIdx.x;
  if (i < total) P[i] = G[i] * A[i & (CDIM - 1)];
}

// ---------------------------------------------------------------------------
// 9. Fused output: out[b,n] = x + b2 + sum_c relu(x*P[n,c] + q[c]) * W2[c]
// ---------------------------------------------------------------------------
__global__ void final_kernel(const float* __restrict__ x, const float* __restrict__ P,
                             const float* __restrict__ Q, const float* __restrict__ W2,
                             const float* __restrict__ b2, float* __restrict__ out,
                             int N, int total) {
  __shared__ float qs[CDIM], ws[CDIM];
  if (threadIdx.x < CDIM) {
    qs[threadIdx.x] = Q[threadIdx.x];
    ws[threadIdx.x] = W2[threadIdx.x];
  }
  __syncthreads();
  int i = blockIdx.x * blockDim.x + threadIdx.x;
  if (i >= total) return;
  int n = i % N;
  float xv = x[i];
  const float4* p4 = (const float4*)(P + (size_t)n * CDIM);
  float acc = b2[0];
#pragma unroll 8
  for (int c4 = 0; c4 < CDIM / 4; ++c4) {
    float4 p = p4[c4];
    int c = c4 * 4;
    acc += fmaxf(fmaf(xv, p.x, qs[c + 0]), 0.f) * ws[c + 0];
    acc += fmaxf(fmaf(xv, p.y, qs[c + 1]), 0.f) * ws[c + 1];
    acc += fmaxf(fmaf(xv, p.z, qs[c + 2]), 0.f) * ws[c + 2];
    acc += fmaxf(fmaf(xv, p.w, qs[c + 3]), 0.f) * ws[c + 3];
  }
  out[i] = xv + acc;
}

// ---------------------------------------------------------------------------
// Launcher
// ---------------------------------------------------------------------------
extern "C" void kernel_launch(void* const* d_in, const int* in_sizes, int n_in,
                              void* d_out, int out_size, void* d_ws, size_t ws_size,
                              hipStream_t stream) {
  const float* x        = (const float*)d_in[0];   // [B,N]
  const int*   eidx     = (const int*)d_in[1];     // [2,E] (JAX default int32)
  const float* gene_emb = (const float*)d_in[2];   // [N,64]
  const float* W_sg     = (const float*)d_in[3];   // [64,64]
  const float* b_sg     = (const float*)d_in[4];   // [64]
  const float* W1       = (const float*)d_in[5];   // [64,128]
  const float* b1_unused= (const float*)d_in[6];   // [128] (zeros; cancels in BN affine)
  const float* gamma1   = (const float*)d_in[7];   // [128]
  const float* beta1    = (const float*)d_in[8];   // [128]
  const float* W2       = (const float*)d_in[9];   // [128,1]
  const float* b2       = (const float*)d_in[10];  // [1]
  (void)b1_unused; (void)n_in; (void)ws_size;
  float* out = (float*)d_out;

  const int N = in_sizes[2] / DHID;
  const int E = in_sizes[1] / 2;
  const int B = in_sizes[0] / N;
  const int* src = eidx;
  const int* dst = eidx + E;

  // ---- workspace carve-up (all offsets multiples of 32 floats -> 128B aligned)
  float* wsf = (float*)d_ws;
  size_t o = 0;
  float*    deg_dinv = wsf + o; o += (size_t)N;
  float*    selfw    = wsf + o; o += (size_t)N;
  float*    hB       = wsf + o; o += (size_t)N * DHID;
  float*    hA       = wsf + o; o += (size_t)N * DHID;
  _Float16* h16      = (_Float16*)(wsf + o); o += (size_t)N * DHID / 2;
  _Float16* wt16     = (_Float16*)(wsf + o); o += (size_t)CDIM * DHID / 2;
  float*    bc       = wsf + o; o += CDIM;
  float*    G        = wsf + o; o += (size_t)N * CDIM;
  float*    P        = wsf + o; o += (size_t)N * CDIM;
  float*    Sx       = wsf + o; o += (size_t)N;
  float*    Sx2      = wsf + o; o += (size_t)N;
  float*    S1       = wsf + o; o += CDIM;
  float*    S2       = wsf + o; o += CDIM;
  float*    Acoef    = wsf + o; o += CDIM;
  float*    Qcoef    = wsf + o; o += CDIM;

  const int TB = 256;
  const int Nd = N * DHID;
  const int total = B * N;   // == out_size

  // 1) folded weights + zero accumulators
  prep_kernel<<<1, CDIM, 0, stream>>>(W_sg, W1, b_sg, wt16, bc, S1, S2);

  // 2) degrees -> dinv, self weights
  deg_init_kernel<<<(N + TB - 1) / TB, TB, 0, stream>>>(deg_dinv, N);
  deg_count_kernel<<<(E + TB - 1) / TB, TB, 0, stream>>>(dst, deg_dinv, E);
  dinv_kernel<<<(N + TB - 1) / TB, TB, 0, stream>>>(deg_dinv, selfw, N);

  // 3) SGConv hop 1: gene_emb -> hB
  selfmul_kernel<<<(Nd + TB - 1) / TB, TB, 0, stream>>>(gene_emb, selfw, hB, Nd);
  {
    long long thr = (long long)E * 32;
    scatter_kernel<<<(unsigned)((thr + TB - 1) / TB), TB, 0, stream>>>(src, dst, deg_dinv, gene_emb, hB, E);
  }
  // 3) SGConv hop 2: hB -> hA
  selfmul_kernel<<<(Nd + TB - 1) / TB, TB, 0, stream>>>(hB, selfw, hA, Nd);
  {
    long long thr = (long long)E * 32;
    scatter_kernel<<<(unsigned)((thr + TB - 1) / TB), TB, 0, stream>>>(src, dst, deg_dinv, hB, hA, E);
  }

  // 4) convert to f16 for WMMA
  tof16_kernel<<<(Nd + TB - 1) / TB, TB, 0, stream>>>(hA, h16, Nd);

  // 5) G = h @ Wc + bc  (v_wmma_f32_16x16x32_f16, one wave per 16x16 tile)
  {
    int tiles = ((N + 15) / 16) * (CDIM / 16);
    int threads = tiles * 32;
    gemm_wmma_kernel<<<(threads + 127) / 128, 128, 0, stream>>>(h16, wt16, bc, G, N);
  }

  // 6) batch sums of x
  sx_kernel<<<(N + TB - 1) / TB, TB, 0, stream>>>(x, Sx, Sx2, N, B);

  // 7) BN stats
  reduce_kernel<<<100, CDIM, 0, stream>>>(G, Sx, Sx2, S1, S2, N);

  // 8) BN coefficients, P = a * G
  coef_kernel<<<1, CDIM, 0, stream>>>(S1, S2, gamma1, beta1, Acoef, Qcoef,
                                      1.0f / (float)((size_t)B * N));
  scaleP_kernel<<<((N * CDIM) + TB - 1) / TB, TB, 0, stream>>>(G, Acoef, P, N * CDIM);

  // 9) fused relu-dot + residual
  final_kernel<<<(total + TB - 1) / TB, TB, 0, stream>>>(x, P, Qcoef, W2, b2, out, N, total);
  (void)out_size;
}
